// N4_48077863911565
// MI455X (gfx1250) — compile-verified, matrix operand order checked by reading, not run on hardware
//
#include <hip/hip_runtime.h>
#include <math.h>

// ---------------------------------------------------------------------------
// Problem constants (B=2, S=2048, E=2048, HEADS=16, d=128)
// ---------------------------------------------------------------------------
#define B_  2
#define S_  2048
#define E_  2048
#define H_  16
#define D_  128
#define N1_ 6160            // 3*E + H
#define NP_ 6272            // padded to multiple of 128
#define M_  (B_ * S_)       // 4096
#define LDP 40              // padded LDS leading dim (elements) -> bank-conflict-free
#define SCALE_ 0.022097086912079608f   // 1/sqrt(E)

typedef __attribute__((ext_vector_type(16))) __bf16 v16bf;
typedef __attribute__((ext_vector_type(8)))  __bf16 v8bf;
typedef __attribute__((ext_vector_type(8)))  float  v8f;

union F8 { v8f v; float f[8]; };

__device__ __forceinline__ __bf16 f2bf(float x) {
  union { float f; unsigned u; } a; a.f = x;
  unsigned r = (a.u + 0x7FFFu + ((a.u >> 16) & 1u)) >> 16;   // round-nearest-even
  union { unsigned short s; __bf16 b; } o; o.s = (unsigned short)r;
  return o.b;
}

// Load one 16x32 bf16 WMMA fragment (A layout; B operands are pre-transposed
// so the same loader applies). p = tile origin in a row-major matrix, ld in
// elements. Lane l: row = l&15; half-wave selects K {0..7,16..23} vs
// {8..15,24..31}; two contiguous 16B loads per lane.
__device__ __forceinline__ v16bf ldfrag(const __bf16* p, int ld) {
  const int lane = threadIdx.x & 31;
  const __bf16* base = p + (size_t)(lane & 15) * ld + ((lane >> 4) << 3);
  v8bf lo = *(const v8bf*)(base);
  v8bf hi = *(const v8bf*)(base + 16);
  return __builtin_shufflevector(lo, hi, 0,1,2,3,4,5,6,7,8,9,10,11,12,13,14,15);
}

__device__ __forceinline__ v8f wmma_bf16(v16bf a, v16bf b, v8f c) {
  return __builtin_amdgcn_wmma_f32_16x16x32_bf16(false, a, false, b,
                                                 (short)0, c, false, false);
}

// CDNA5 async global->LDS copy (ASYNCcnt path, CDNA5 ISA).
// lds_off = low 32 bits of the generic LDS address (== wave-relative LDS byte
// address per the aperture truncation rule); gptr = per-lane global address.
__device__ __forceinline__ void async_ld16(unsigned lds_off, const void* gptr) {
  asm volatile("global_load_async_to_lds_b128 %0, %1, off"
               :: "v"(lds_off), "v"(gptr) : "memory");
}
__device__ __forceinline__ void wait_async0() {
  asm volatile("s_wait_asynccnt 0x0" ::: "memory");
}

// ---------------------------------------------------------------------------
// Conversion / prep kernels
// ---------------------------------------------------------------------------
__global__ void k_cvt_x(const float* __restrict__ x, __bf16* __restrict__ xb, int n) {
  int i = blockIdx.x * blockDim.x + threadIdx.x;
  if (i < n) xb[i] = f2bf(x[i]);
}

// W_qkvh (E x N1) -> bf16 transposed + zero padded (NP x E)
__global__ void k_cvt_wq(const float* __restrict__ w, __bf16* __restrict__ wt) {
  int i = blockIdx.x * blockDim.x + threadIdx.x;
  if (i >= NP_ * E_) return;
  int n = i / E_, k = i % E_;
  float v = (n < N1_) ? w[(size_t)k * N1_ + n] : 0.0f;
  wt[i] = f2bf(v);
}

// W_out (E x E) -> bf16 transposed (E x E)
__global__ void k_cvt_wout(const float* __restrict__ w, __bf16* __restrict__ wt) {
  int i = blockIdx.x * blockDim.x + threadIdx.x;
  if (i >= E_ * E_) return;
  int n = i / E_, k = i % E_;
  wt[i] = f2bf(w[(size_t)k * E_ + n]);
}

// ---------------------------------------------------------------------------
// WMMA GEMM with async double-buffered LDS staging.
// C(MxN,f32) = A(MxK,bf16 row-major) * Bt(NxK,bf16 row-major)^T
// block = 256 threads = 8 waves; block tile 128x128; wave tile 64x32 (4x2 WMMA)
// LDS: 2 buffers x (A 128xLDP + B 128xLDP) bf16 = 40KB.
// Each thread async-copies one 64B row slice (4 x b128) per k-step.
// ---------------------------------------------------------------------------
__global__ __launch_bounds__(256) void k_gemm_lds(
    const __bf16* __restrict__ A, const __bf16* __restrict__ Bt,
    float* __restrict__ C, int M, int N, int K, int ldc) {
  __shared__ __align__(16) __bf16 lds[2][2][128 * LDP];

  const int tid  = threadIdx.x;
  const int w    = tid >> 5;
  const int lane = tid & 31;
  const int m0   = blockIdx.y * 128;
  const int n0   = blockIdx.x * 128;

  // --- loader mapping: 256 threads cover 128 A-rows + 128 B-rows
  const int lrow = tid & 127;
  const int lmat = tid >> 7;                 // 0 = A, 1 = Bt
  const __bf16* gsrc = (lmat == 0) ? (A  + (size_t)(m0 + lrow) * K)
                                   : (Bt + (size_t)(n0 + lrow) * K);
  unsigned lb0 = (unsigned)(uintptr_t)(const void*)&lds[0][lmat][lrow * LDP];
  unsigned lb1 = (unsigned)(uintptr_t)(const void*)&lds[1][lmat][lrow * LDP];

  // --- compute mapping: wave grid 2x4 -> wave tile 64 rows x 32 cols
  const int wm = (w >> 2) * 64;
  const int wn = (w & 3) * 32;
  const int colb = lane & 15, hi16 = lane >> 4;

  F8 acc[4][2];
  for (int r = 0; r < 4; ++r) { acc[r][0].v = {}; acc[r][1].v = {}; }

  const int nk = K >> 5;
  // prologue: fill buffer 0
  {
    const __bf16* g = gsrc;
    async_ld16(lb0 +  0, g);
    async_ld16(lb0 + 16, g + 8);
    async_ld16(lb0 + 32, g + 16);
    async_ld16(lb0 + 48, g + 24);
  }
  for (int i = 0; i < nk; ++i) {
    const int buf = i & 1;
    wait_async0();
    __syncthreads();                         // LDS fill visible to all waves
    if (i + 1 < nk) {                        // prefetch next tile into back buffer
      const __bf16* g = gsrc + (i + 1) * 32;
      unsigned lb = (buf == 0) ? lb1 : lb0;
      async_ld16(lb +  0, g);
      async_ld16(lb + 16, g + 8);
      async_ld16(lb + 32, g + 16);
      async_ld16(lb + 48, g + 24);
    }
    const __bf16* la = &lds[buf][0][0];
    const __bf16* lb = &lds[buf][1][0];
    v16bf bf0 = ldfrag(lb + (wn +  0) * LDP, LDP);
    v16bf bf1 = ldfrag(lb + (wn + 16) * LDP, LDP);
    for (int r = 0; r < 4; ++r) {
      v16bf af = ldfrag(la + (wm + r * 16) * LDP, LDP);
      acc[r][0].v = wmma_bf16(af, bf0, acc[r][0].v);
      acc[r][1].v = wmma_bf16(af, bf1, acc[r][1].v);
    }
    // per-wave DS reads complete before its WMMAs consume them; the next
    // iteration's leading barrier orders them against back-buffer refill.
  }

  for (int r = 0; r < 4; ++r)
    for (int c = 0; c < 2; ++c)
      for (int i = 0; i < 8; ++i) {
        int row = m0 + wm + r * 16 + i + 8 * hi16;
        int col = n0 + wn + c * 16 + colb;
        C[(size_t)row * ldc + col] = acc[r][c].f[i];
      }
}

// ---------------------------------------------------------------------------
// Split Q and V^T out of proj into bf16 head-major tensors
// Q: (B,H,S,D)  Vt: (B,H,D,S)
// ---------------------------------------------------------------------------
__global__ void k_qv(const float* __restrict__ proj,
                     __bf16* __restrict__ Qb, __bf16* __restrict__ Vtb) {
  int i = blockIdx.x * blockDim.x + threadIdx.x;
  if (i >= B_ * H_ * S_ * D_) return;
  int d = i & (D_ - 1);
  int s = (i >> 7) & (S_ - 1);
  int h = (i >> 18) & (H_ - 1);
  int b = i >> 22;
  size_t row = (size_t)(b * S_ + s) * NP_;
  Qb[(((size_t)(b * H_ + h)) * S_ + s) * D_ + d]  = f2bf(proj[row + h * 384 + d]);
  Vtb[(((size_t)(b * H_ + h)) * D_ + d) * S_ + s] = f2bf(proj[row + h * 384 + 256 + d]);
}

// Gated cumulative keys: Kc[s] = sum_{t<=s} (t+1)*tanh(g[t])*K[t] / (s+1)^2
// one thread per (b,h,d) channel (d contiguous -> coalesced), sequential in s.
__global__ void k_scan(const float* __restrict__ proj, __bf16* __restrict__ Kcb) {
  int t = blockIdx.x * blockDim.x + threadIdx.x;
  if (t >= B_ * H_ * D_) return;
  int d = t & (D_ - 1);
  int h = (t >> 7) & (H_ - 1);
  int b = t >> 11;
  float acc = 0.0f;
  for (int s = 0; s < S_; ++s) {
    size_t row = (size_t)(b * S_ + s) * NP_;
    float kv = proj[row + h * 384 + 128 + d];
    float g  = tanhf(proj[row + 3 * E_ + h]);
    acc += (float)(s + 1) * g * kv;
    float inv = 1.0f / ((float)(s + 1) * (float)(s + 1));
    Kcb[(((size_t)(b * H_ + h)) * S_ + s) * D_ + d] = f2bf(acc * inv);
  }
}

// ---------------------------------------------------------------------------
// Flash attention: one wave per (b,h,16-row q tile); 32-key tiles.
// scores: 8 WMMAs / tile, PV: 8 WMMAs / tile. Online softmax with
// cross-lane shuffles; P converted to bf16 A-fragment via 1KB LDS.
// ---------------------------------------------------------------------------
__global__ __launch_bounds__(32) void k_attn(
    const __bf16* __restrict__ Qb, const __bf16* __restrict__ Kcb,
    const __bf16* __restrict__ Vtb, __bf16* __restrict__ Ob) {
  const int q0 = blockIdx.x * 16;
  const int h  = blockIdx.y;
  const int b  = blockIdx.z;
  const int lane = threadIdx.x & 31;
  const int colb = lane & 15, hi16 = lane >> 4;

  __shared__ __align__(16) __bf16 Pl[16 * 32];

  const __bf16* Qh  = Qb  + ((size_t)(b * H_ + h) * S_) * D_;
  const __bf16* Kch = Kcb + ((size_t)(b * H_ + h) * S_) * D_;
  const __bf16* Vth = Vtb + ((size_t)(b * H_ + h) * D_) * S_;

  v16bf qf[4];
  for (int kk = 0; kk < 4; ++kk)
    qf[kk] = ldfrag(Qh + (size_t)q0 * D_ + kk * 32, D_);

  F8 acc_o[8];
  for (int t = 0; t < 8; ++t) acc_o[t].v = {};
  float m_i[8], l_i[8];
  for (int i = 0; i < 8; ++i) { m_i[i] = -INFINITY; l_i[i] = 0.0f; }

  const int njt = (q0 >> 5) + 1;
  for (int jt = 0; jt < njt; ++jt) {
    const int kbase = jt * 32;
    F8 s0, s1; s0.v = {}; s1.v = {};
    for (int kk = 0; kk < 4; ++kk) {
      v16bf kb0 = ldfrag(Kch + (size_t)(kbase)      * D_ + kk * 32, D_);
      v16bf kb1 = ldfrag(Kch + (size_t)(kbase + 16) * D_ + kk * 32, D_);
      s0.v = wmma_bf16(qf[kk], kb0, s0.v);
      s1.v = wmma_bf16(qf[kk], kb1, s1.v);
    }
    const bool diag = (jt == njt - 1);
    for (int i = 0; i < 8; ++i) {
      s0.f[i] *= SCALE_; s1.f[i] *= SCALE_;
      if (diag) {
        int row = q0 + i + 8 * hi16;
        if (kbase + colb      > row) s0.f[i] = -INFINITY;
        if (kbase + 16 + colb > row) s1.f[i] = -INFINITY;
      }
    }
    float sc[8];
    for (int i = 0; i < 8; ++i) {
      float rm = fmaxf(s0.f[i], s1.f[i]);
      for (int off = 1; off < 16; off <<= 1)
        rm = fmaxf(rm, __shfl_xor(rm, off, 32));
      float mn = fmaxf(m_i[i], rm);
      sc[i] = __expf(m_i[i] - mn);
      m_i[i] = mn;
      s0.f[i] = __expf(s0.f[i] - mn);
      s1.f[i] = __expf(s1.f[i] - mn);
      float rs = s0.f[i] + s1.f[i];
      for (int off = 1; off < 16; off <<= 1)
        rs += __shfl_xor(rs, off, 32);
      l_i[i] = l_i[i] * sc[i] + rs;
    }
    for (int t = 0; t < 8; ++t)
      for (int i = 0; i < 8; ++i) acc_o[t].f[i] *= sc[i];

    for (int i = 0; i < 8; ++i) {
      Pl[(i + 8 * hi16) * 32 + colb]      = f2bf(s0.f[i]);
      Pl[(i + 8 * hi16) * 32 + 16 + colb] = f2bf(s1.f[i]);
    }
    __syncthreads();
    v16bf pa = ldfrag(Pl, 32);
    for (int t = 0; t < 8; ++t) {
      v16bf vb = ldfrag(Vth + (size_t)(t * 16) * S_ + kbase, S_);
      acc_o[t].v = wmma_bf16(pa, vb, acc_o[t].v);
    }
    __syncthreads();
  }

  for (int t = 0; t < 8; ++t)
    for (int i = 0; i < 8; ++i) {
      int row = q0 + i + 8 * hi16;
      float o = acc_o[t].f[i] / l_i[i];
      Ob[((size_t)(b * S_ + row)) * E_ + h * D_ + t * 16 + colb] = f2bf(o);
    }
}

// ---------------------------------------------------------------------------
// Host launcher
// ---------------------------------------------------------------------------
extern "C" void kernel_launch(void* const* d_in, const int* in_sizes, int n_in,
                              void* d_out, int out_size, void* d_ws, size_t ws_size,
                              hipStream_t stream) {
  const float* x     = (const float*)d_in[0];
  const float* Wqkvh = (const float*)d_in[1];
  const float* Wout  = (const float*)d_in[2];
  float* out = (float*)d_out;

  char* ws = (char*)d_ws;
  const size_t SZ_XB   = (size_t)M_ * E_ * 2;
  const size_t SZ_WQT  = (size_t)NP_ * E_ * 2;
  const size_t SZ_PROJ = (size_t)M_ * NP_ * 4;
  const size_t SZ_HEAD = (size_t)B_ * H_ * S_ * D_ * 2;

  __bf16* xb   = (__bf16*)(ws);
  __bf16* wqT  = (__bf16*)(ws + SZ_XB);
  float*  proj = (float*) (ws + SZ_XB + SZ_WQT);
  __bf16* Qb   = (__bf16*)(ws + SZ_XB + SZ_WQT + SZ_PROJ);
  __bf16* Kcb  = (__bf16*)(ws + SZ_XB + SZ_WQT + SZ_PROJ + SZ_HEAD);
  __bf16* Vtb  = (__bf16*)(ws + SZ_XB + SZ_WQT + SZ_PROJ + 2 * SZ_HEAD);
  __bf16* Ob   = (__bf16*)(ws + SZ_XB + SZ_WQT + SZ_PROJ + 3 * SZ_HEAD);
  __bf16* woT  = (__bf16*)(ws + SZ_XB + SZ_WQT + SZ_PROJ + 3 * SZ_HEAD + SZ_XB);

  // 1) conversions
  {
    int n = M_ * E_;
    k_cvt_x<<<(n + 255) / 256, 256, 0, stream>>>(x, xb, n);
  }
  k_cvt_wq  <<<(NP_ * E_ + 255) / 256, 256, 0, stream>>>(Wqkvh, wqT);
  k_cvt_wout<<<(E_ * E_ + 255) / 256, 256, 0, stream>>>(Wout, woT);

  // 2) proj = x @ W_qkvh   (4096 x 6272, K=2048)
  k_gemm_lds<<<dim3(NP_ / 128, M_ / 128), 256, 0, stream>>>(
      xb, wqT, proj, M_, NP_, E_, NP_);

  // 3) head split + gated key scan
  {
    int n = B_ * H_ * S_ * D_;
    k_qv<<<(n + 255) / 256, 256, 0, stream>>>(proj, Qb, Vtb);
  }
  k_scan<<<(B_ * H_ * D_ + 255) / 256, 256, 0, stream>>>(proj, Kcb);

  // 4) flash attention
  k_attn<<<dim3(S_ / 16, H_, B_), 32, 0, stream>>>(Qb, Kcb, Vtb, Ob);

  // 5) out = attn_out @ W_out  -> d_out fp32
  k_gemm_lds<<<dim3(E_ / 128, M_ / 128), 256, 0, stream>>>(
      Ob, woT, out, M_, E_, E_, E_);

  (void)in_sizes; (void)n_in; (void)out_size; (void)ws_size;
}
